// SIRENdecoder_16836271800898
// MI455X (gfx1250) — matrix operand order repaired and checked by reading, not compiled, not run on hardware
//
#include <hip/hip_runtime.h>

typedef __attribute__((ext_vector_type(16))) __bf16 v16bf;
typedef __attribute__((ext_vector_type(8)))  __bf16 v8bf;
typedef __attribute__((ext_vector_type(8)))  float  v8f;
typedef __attribute__((ext_vector_type(4)))  unsigned int u32x4;
typedef __attribute__((ext_vector_type(8)))  int i32x8;
typedef __attribute__((ext_vector_type(4)))  int i32x4;

#define T_DIM   8
#define N_PTS   32768
#define RANKV   32
#define W0F     30.0f
#define TILE    128         // tokens per block (8 waves x 16 rows)
#define NTHR    256
#define XPAD    264         // 256 + 8 bf16 -> 528B row stride (16B aligned)

#if __has_builtin(__builtin_amdgcn_tensor_load_to_lds)
#define HAVE_TDM 1
#else
#define HAVE_TDM 0
#endif
#if __has_builtin(__builtin_amdgcn_global_load_async_to_lds_b128)
#define HAVE_ASYNC 1
#else
#define HAVE_ASYNC 0
#endif

typedef __attribute__((address_space(1))) void g1void;
typedef __attribute__((address_space(3))) void l3void;

__device__ __forceinline__ float softplusf(float x) {
  return (x > 20.f) ? x : log1pf(__expf(x));
}

// ---------------------------------------------------------------------------
// Hypernetwork: W_eff[i][o] = w[o][i]*scale_w(o) + sum_r A[t,i,r]*B[t,r,o]
// (bf16) plus fused bias b[o] + hyper_bias[o] (f32). Layer 0 K-rows 3..255 = 0.
// ---------------------------------------------------------------------------
__global__ __launch_bounds__(256) void hyper_kernel(
    const float* __restrict__ latents,
    const float* __restrict__ nf_w_first, const float* __restrict__ nf_b_first,
    const float* __restrict__ nf_c_first,
    const float* __restrict__ nf_w_hid,   const float* __restrict__ nf_b_hid,
    const float* __restrict__ nf_c_hid,
    const float* __restrict__ Aw_first,   const float* __restrict__ Ac_first,
    const float* __restrict__ Aw_hid,     const float* __restrict__ Ac_hid,
    const float* __restrict__ Bw,         const float* __restrict__ Bc,
    const float* __restrict__ hbw,        const float* __restrict__ hbc,
    __bf16* __restrict__ Weff, float* __restrict__ biasW)
{
  const int l = blockIdx.x >> 3;
  const int t = blockIdx.x & 7;
  const int tid = threadIdx.x;

  const int d_in = (l == 0) ? 3 : 256;
  const float* w  = (l == 0) ? nf_w_first : nf_w_hid + (size_t)(l - 1) * 256 * 256;
  const float* b  = (l == 0) ? nf_b_first : nf_b_hid + (size_t)(l - 1) * 256;
  const float  c  = (l == 0) ? nf_c_first[0] : nf_c_hid[l - 1];
  const float* aw = (l == 0) ? Aw_first : Aw_hid + (size_t)(l - 1) * 8192 * 128;
  const float  ac = (l == 0) ? Ac_first[0] : Ac_hid[l - 1];
  const float* bw = Bw + (size_t)l * 8192 * 128;
  const float  bc = Bc[l];
  const float* hw = hbw + (size_t)l * 256 * 128;
  const float  hc = hbc[l];

  __shared__ float  lat_s[128];
  __shared__ __bf16 A_ld[8192];   // A[t, i, r] at i*32+r
  __shared__ __bf16 B_ld[8192];   // B[t, r, o] at r*256+o

  if (tid < 128) lat_s[tid] = latents[t * 128 + tid];
  __syncthreads();

  const float spa = softplusf(ac);
  const float spb = softplusf(bc);
  const float sph = softplusf(hc);
  const float spw = softplusf(c);

  for (int row = tid; row < d_in * RANKV; row += 256) {
    const float* ar = aw + (size_t)row * 128;
    float l1 = 0.f, dot = 0.f;
    for (int k = 0; k < 128; ++k) { float v = ar[k]; l1 += fabsf(v); dot += lat_s[k] * v; }
    A_ld[row] = (__bf16)(dot * fminf(spa / l1, 1.f));
  }
  for (int row = tid; row < RANKV * 256; row += 256) {
    const float* br = bw + (size_t)row * 128;
    float l1 = 0.f, dot = 0.f;
    for (int k = 0; k < 128; ++k) { float v = br[k]; l1 += fabsf(v); dot += lat_s[k] * v; }
    B_ld[row] = (__bf16)(dot * fminf(spb / l1, 1.f));
  }
  {
    const int o = tid;
    const float* hr = hw + (size_t)o * 128;
    float l1 = 0.f, dot = 0.f;
    for (int k = 0; k < 128; ++k) { float v = hr[k]; l1 += fabsf(v); dot += lat_s[k] * v; }
    biasW[(size_t)(l * 8 + t) * 256 + o] = dot * fminf(sph / l1, 1.f) + b[o];
  }
  __syncthreads();

  {
    const int o = tid;
    const float* wr = w + (size_t)o * d_in;
    float l1 = 0.f;
    for (int i = 0; i < d_in; ++i) l1 += fabsf(wr[i]);
    const float sw = fminf(spw / l1, 1.f);
    const size_t base = (size_t)(l * 8 + t) * 256 * 256;
    for (int i = 0; i < d_in; ++i) {
      float acc = wr[i] * sw;
      #pragma unroll 8
      for (int r = 0; r < RANKV; ++r)
        acc += (float)A_ld[i * RANKV + r] * (float)B_ld[r * 256 + o];
      Weff[base + (size_t)i * 256 + o] = (__bf16)acc;
    }
    for (int i = d_in; i < 256; ++i)
      Weff[base + (size_t)i * 256 + o] = (__bf16)0.f;
  }
}

// ---------------------------------------------------------------------------
__global__ __launch_bounds__(256) void lastprep_kernel(
    const float* __restrict__ w, const float* __restrict__ b,
    const float* __restrict__ c, float* __restrict__ wl, float* __restrict__ blast)
{
  __shared__ float red[256];
  const int tid = threadIdx.x;
  red[tid] = fabsf(w[tid]);
  __syncthreads();
  for (int s = 128; s > 0; s >>= 1) {
    if (tid < s) red[tid] += red[tid + s];
    __syncthreads();
  }
  const float scale = fminf(softplusf(c[0]) / red[0], 1.f);
  wl[tid] = w[tid] * scale;
  if (tid == 0) blast[0] = b[0];
}

// ---------------------------------------------------------------------------
// Weight-chunk staging: TDM tensor_load_to_lds (preferred) -> async
// global->LDS (fallback) -> synchronous copy (last resort).
// Chunk = 32 K-rows x 256 N (bf16, 16 KB), LDS rows padded to 264 elements
// (TDM pad: 4 DWORDs after every 128 DWORDs).
// ---------------------------------------------------------------------------
__device__ __forceinline__ void stage_chunk(const __bf16* __restrict__ Wg, int kc,
                                            __bf16 (*buf)[XPAD], int tid) {
#if HAVE_TDM
  if (tid < 32) {  // one TDM op per block, issued by wave 0 (TDM ignores EXEC)
    const unsigned long long ga = (unsigned long long)(const void*)(Wg + (size_t)kc * 256);
    const unsigned lds = (unsigned)(unsigned long long)(l3void*)&buf[0][0];
    // D# group 0: count=1, lds_addr, global_addr, type=2
    u32x4 g0 = { 0x1u, lds, (unsigned)ga, (unsigned)(ga >> 32) | 0x80000000u };
    // D# group 1: data_size=2B, pad_enable, pad_interval=128 DW, pad_amount=4 DW,
    // tensor_dim0=256, tensor_dim1=256, tile_dim0=256, tile_dim1=32, dim0_stride=256
    i32x8 g1 = { 0x07910000, 0x01000000, 0x01000000, 0x01000000, 32, 256, 0, 0 };
    i32x4 z4 = { 0, 0, 0, 0 };
#if defined(__clang_major__) && (__clang_major__ >= 23)
    i32x8 z8 = { 0, 0, 0, 0, 0, 0, 0, 0 };
    __builtin_amdgcn_tensor_load_to_lds(g0, g1, z4, z4, z8, 0);
#else
    __builtin_amdgcn_tensor_load_to_lds(g0, g1, z4, z4, 0);
#endif
  }
#elif HAVE_ASYNC
  for (int e = tid; e < 1024; e += NTHR) {
    const int r = e >> 5, s = e & 31;
    __builtin_amdgcn_global_load_async_to_lds_b128(
        (g1void*)(unsigned long long)(const void*)(Wg + (size_t)(kc + r) * 256 + s * 8),
        (l3void*)&buf[r][s * 8], 0, 0);
  }
#else
  for (int e = tid; e < 1024; e += NTHR) {
    const int r = e >> 5, s = e & 31;
    *(v8bf*)&buf[r][s * 8] = *(const v8bf*)&Wg[(size_t)(kc + r) * 256 + s * 8];
  }
#endif
}

__device__ __forceinline__ void stage_wait(int tid) {
#if HAVE_TDM
  if (tid < 32) {
#if __has_builtin(__builtin_amdgcn_s_wait_tensorcnt)
    __builtin_amdgcn_s_wait_tensorcnt(0);
#else
    asm volatile("s_wait_tensorcnt 0x0" ::: "memory");
#endif
  }
#elif HAVE_ASYNC
#if __has_builtin(__builtin_amdgcn_s_wait_asynccnt)
  __builtin_amdgcn_s_wait_asynccnt(0);
#else
  asm volatile("s_wait_asynccnt 0x0" ::: "memory");
#endif
#else
  (void)tid;
#endif
}

// B fragment: lane%16 = N column, 16 K values per lane (k = 8h+s, 16+8h+s-8)
__device__ __forceinline__ v16bf loadB(const __bf16 (*wb)[XPAD], int lane, int nt) {
  const int n = nt * 16 + (lane & 15);
  const int h8 = (lane >> 4) * 8;
  v16bf r;
  #pragma unroll
  for (int s = 0; s < 8; ++s) {
    r[s] = wb[h8 + s][n];
    r[s + 8] = wb[16 + h8 + s][n];
  }
  return r;
}

// ---------------------------------------------------------------------------
// Fused SIREN stack: 128-token tile resident in LDS (bf16), 8 waves each
// owning 16 token rows. Weight chunks double-buffered ACROSS layers so one
// TDM transfer is always in flight (it also overlaps the sin epilogue).
// B fragments are register-pipelined (depth 2) to hide ds_load latency.
// ---------------------------------------------------------------------------
__global__ __launch_bounds__(NTHR) void siren_main(
    const float* __restrict__ coords,
    const __bf16* __restrict__ Weff,
    const float* __restrict__ biasW,
    const float* __restrict__ wl,
    const float* __restrict__ blast,
    float* __restrict__ out)
{
  __shared__ __attribute__((aligned(16))) __bf16 xs[TILE][XPAD];    // activations
  __shared__ __attribute__((aligned(16))) __bf16 wsb[2][32][XPAD];  // staged W_eff chunks

  const int tid  = threadIdx.x;
  const int wave = tid >> 5;
  const int lane = tid & 31;
  const int lm   = lane & 15;
  const bool hi  = lane >= 16;

  const int t    = blockIdx.x >> 8;     // 256 tiles per task
  const int tile = blockIdx.x & 255;
  const int tok0 = tile * TILE;

  // coords into xs[:, 0..2], zero-pad cols 3..31 (layer-0 K = 32)
  const float* cbase = coords + ((size_t)t * N_PTS + tok0) * 3;
  for (int e = tid; e < TILE * 32; e += NTHR) {
    const int r = e >> 5, cc = e & 31;
    const float v = (cc < 3) ? cbase[(size_t)r * 3 + cc] : 0.f;
    xs[r][cc] = (__bf16)v;
  }
  __syncthreads();

  // prime the pipeline: layer-0 chunk 0 -> wsb[0]
  stage_chunk(Weff + (size_t)t * 65536, 0, wsb[0], tid);
  int bufi = 0;

  for (int l = 0; l < 4; ++l) {
    const int nch = (l == 0) ? 1 : 8;
    const __bf16* Wg = Weff + (size_t)(l * 8 + t) * 65536;
    const float* bptr = biasW + (size_t)(l * 8 + t) * 256;
    __builtin_prefetch(bptr, 0, 0);     // global_prefetch for epilogue bias row

    v8f acc[16] = {};

    for (int ic = 0; ic < nch; ++ic) {
      stage_wait(tid);                  // chunk ic landed in wsb[bufi]
      __syncthreads();                  // publish; everyone past wsb[bufi^1]
      // look-ahead: next chunk of this layer, or first chunk of next layer
      if (ic + 1 < nch)
        stage_chunk(Wg, ic + 1, wsb[bufi ^ 1], tid);
      else if (l + 1 < 4)
        stage_chunk(Weff + (size_t)((l + 1) * 8 + t) * 65536, 0, wsb[bufi ^ 1], tid);

      const int kc = ic * 32;
      // A fragment: this wave's 16 token rows (ISA 16-bit A layout)
      const int arow = wave * 16 + lm;
      const int o1 = kc + (hi ? 8 : 0);
      const int o2 = kc + (hi ? 24 : 16);
      const v8bf alo = *(const v8bf*)&xs[arow][o1];
      const v8bf ahi = *(const v8bf*)&xs[arow][o2];
      const v16bf afrag = __builtin_shufflevector(
          alo, ahi, 0, 1, 2, 3, 4, 5, 6, 7, 8, 9, 10, 11, 12, 13, 14, 15);

      const __bf16 (*wb)[XPAD] = wsb[bufi];
      v16bf b0 = loadB(wb, lane, 0);
      v16bf b1 = loadB(wb, lane, 1);
      #pragma unroll
      for (int nt = 0; nt < 16; ++nt) {
        const v16bf bcur = (nt & 1) ? b1 : b0;
        if (nt + 2 < 16) {               // issue load 2 tiles ahead
          if (nt & 1) b1 = loadB(wb, lane, nt + 2);
          else        b0 = loadB(wb, lane, nt + 2);
        }
        acc[nt] = __builtin_amdgcn_wmma_f32_16x16x32_bf16(
            false, afrag, false, bcur, (short)0, acc[nt], false, false);
      }
      bufi ^= 1;
    }

    // epilogue: x = sin(W0 * (acc + bias)); waves write only their own rows,
    // overlapping the TDM transfer of the next layer's first chunk.
    #pragma unroll
    for (int nt = 0; nt < 16; ++nt) {
      const int nglob = nt * 16 + lm;
      const float bv = bptr[nglob];
      #pragma unroll
      for (int r = 0; r < 8; ++r) {
        const int m = hi ? (r + 8) : r;      // ISA 32-bit C/D layout
        const float v = __sinf(W0F * (acc[nt][r] + bv));
        xs[wave * 16 + m][nglob] = (__bf16)v;
      }
    }
  }
  __syncthreads();

  // head: out = xs @ wl + blast
  const float bl = blast[0];
  for (int tk = tid; tk < TILE; tk += NTHR) {
    float s = bl;
    #pragma unroll 8
    for (int k = 0; k < 256; ++k) s += (float)xs[tk][k] * wl[k];
    out[(size_t)t * N_PTS + tok0 + tk] = s;
  }
}

// ---------------------------------------------------------------------------
extern "C" void kernel_launch(void* const* d_in, const int* in_sizes, int n_in,
                              void* d_out, int out_size, void* d_ws, size_t ws_size,
                              hipStream_t stream) {
  (void)in_sizes; (void)n_in; (void)out_size; (void)ws_size;

  const float* coords     = (const float*)d_in[0];
  const float* latents    = (const float*)d_in[1];
  const float* nf_w_first = (const float*)d_in[2];
  const float* nf_b_first = (const float*)d_in[3];
  const float* nf_c_first = (const float*)d_in[4];
  const float* nf_w_hid   = (const float*)d_in[5];
  const float* nf_b_hid   = (const float*)d_in[6];
  const float* nf_c_hid   = (const float*)d_in[7];
  const float* nf_w_last  = (const float*)d_in[8];
  const float* nf_b_last  = (const float*)d_in[9];
  const float* nf_c_last  = (const float*)d_in[10];
  const float* Aw_first   = (const float*)d_in[11];
  const float* Ac_first   = (const float*)d_in[12];
  const float* Aw_hid     = (const float*)d_in[13];
  const float* Ac_hid     = (const float*)d_in[14];
  const float* Bw         = (const float*)d_in[15];
  const float* Bc         = (const float*)d_in[16];
  const float* hbw        = (const float*)d_in[17];
  const float* hbc        = (const float*)d_in[18];
  float* out = (float*)d_out;

  __bf16* Weff  = (__bf16*)d_ws;                                         // 4 MB
  float*  biasW = (float*)((char*)d_ws + (size_t)4 * 8 * 256 * 256 * 2); // 32 KB
  float*  wl    = biasW + 4 * 8 * 256;
  float*  blast = wl + 256;

  hipLaunchKernelGGL(hyper_kernel, dim3(32), dim3(256), 0, stream,
                     latents, nf_w_first, nf_b_first, nf_c_first,
                     nf_w_hid, nf_b_hid, nf_c_hid,
                     Aw_first, Ac_first, Aw_hid, Ac_hid,
                     Bw, Bc, hbw, hbc, Weff, biasW);

  hipLaunchKernelGGL(lastprep_kernel, dim3(1), dim3(256), 0, stream,
                     nf_w_last, nf_b_last, nf_c_last, wl, blast);

  hipLaunchKernelGGL(siren_main, dim3(T_DIM * (N_PTS / TILE)), dim3(NTHR), 0, stream,
                     coords, Weff, biasW, wl, blast, out);
}
